// CrossBatchEmbeddingMixer_20933670601092
// MI455X (gfx1250) — compile-verified, matrix operand order checked by reading, not run on hardware
//
#include <hip/hip_runtime.h>
#include <stdint.h>

// ---------------------------------------------------------------------------
// CrossBatchEmbeddingMixer for MI455X (gfx1250).
// GEMMs: v_wmma_f32_16x16x32_bf16 (bf16 operands, f32 accum), fed by the
// CDNA5 async memory path (global_load_async_to_lds_b128 + ASYNCcnt) with
// double-buffered LDS tiles. Per K-stage all 12 LDS fragments are preloaded
// into distinct VGPRs so the 16 WMMAs issue as one uninterrupted burst.
// ---------------------------------------------------------------------------

#define B_DIM 4096
#define H_DIM 4096
#define GH_DIM 1024
#define K3H   (3 * H_DIM)

typedef __attribute__((ext_vector_type(16))) __bf16 v16bf;
typedef __attribute__((ext_vector_type(8)))  float  v8f;

union FragBF { uint4 q[2]; v16bf v; };

__device__ __forceinline__ unsigned short f2bf(float f) {
  unsigned int u = __float_as_uint(f);
  u += 0x7FFFu + ((u >> 16) & 1u);      // round-to-nearest-even
  return (unsigned short)(u >> 16);
}

// CDNA5 async copy: memory -> LDS direct, tracked by ASYNCcnt (no VGPR hop).
__device__ __forceinline__ void async_b128(unsigned lds_off, const void* gaddr) {
  asm volatile("global_load_async_to_lds_b128 %0, %1, off"
               :: "v"(lds_off), "v"(gaddr) : "memory");
}
__device__ __forceinline__ void wait_async0() {
  asm volatile("s_wait_asynccnt 0x0" ::: "memory");
}

// ---------------------------------------------------------------------------
// f32 -> bf16 conversion (vectorized x4)
// ---------------------------------------------------------------------------
__global__ void cvt_bf16_kernel(const float* __restrict__ in,
                                unsigned short* __restrict__ out, long n4) {
  long i = (long)blockIdx.x * blockDim.x + threadIdx.x;
  if (i < n4) {
    float4 f = ((const float4*)in)[i];
    uint2 p;
    p.x = (unsigned)f2bf(f.x) | ((unsigned)f2bf(f.y) << 16);
    p.y = (unsigned)f2bf(f.z) | ((unsigned)f2bf(f.w) << 16);
    ((uint2*)out)[i] = p;
  }
}

// ---------------------------------------------------------------------------
// Generic NT GEMM: C[M,N] = A[M,K] * Bt[N,K]^T  (A, Bt row-major bf16)
// 128x128 block tile, BK=64, 256 threads = 8 waves, wave tile 64x32 (4x2 WMMA).
// Double-buffered LDS staged with global_load_async_to_lds_b128.
// Epilogues:
//   0: store f32            1: store bf16 transposed (C[col*ldt+row])
//   2: bias + exact GELU -> bf16
//   3: sigmoid(acc+bias) gate: out = hsrc + gate * csrc   (f32 out)
// ---------------------------------------------------------------------------
enum { EPI_F32 = 0, EPI_BF16T = 1, EPI_GELU = 2, EPI_FINAL = 3 };

template <int EPI>
__global__ __launch_bounds__(256)
void gemm_bf16_nt(const unsigned short* __restrict__ A,
                  const unsigned short* __restrict__ Bt,
                  void* __restrict__ C,
                  int M, int N, int K,
                  const float* __restrict__ bias,
                  const float* __restrict__ hsrc,
                  const float* __restrict__ csrc,
                  int ldt) {
  constexpr int BK  = 64;
  constexpr int LDA = BK + 8;                 // 72 halves = 144B, 16B-aligned rows
  __shared__ __align__(16) unsigned short sA[2][128 * LDA];
  __shared__ __align__(16) unsigned short sB[2][128 * LDA];

  const int tid  = threadIdx.x;
  const int lane = tid & 31;
  const int wid  = tid >> 5;
  const int wm   = wid & 1;                   // 2 wave-rows of 64
  const int wn   = wid >> 1;                  // 4 wave-cols of 32
  const int r    = lane & 15;
  const int hi   = lane >> 4;

  const int bm = blockIdx.y * 128;
  const int bn = blockIdx.x * 128;

  // Per-thread chunk coordinates: 1024 chunks of 8 halves per 128x64 tile.
  int rowc[4], colc[4];
#pragma unroll
  for (int t = 0; t < 4; t++) {
    int chunk = tid + t * 256;
    rowc[t]   = chunk >> 3;
    colc[t]   = (chunk & 7) * 8;
  }

  auto issue_tile = [&](int k0, int buf) {
#pragma unroll
    for (int t = 0; t < 4; t++) {
      const unsigned short* ga = A  + (size_t)(bm + rowc[t]) * K + k0 + colc[t];
      const unsigned short* gb = Bt + (size_t)(bn + rowc[t]) * K + k0 + colc[t];
      unsigned la = (unsigned)(uintptr_t)&sA[buf][rowc[t] * LDA + colc[t]];
      unsigned lb = (unsigned)(uintptr_t)&sB[buf][rowc[t] * LDA + colc[t]];
      async_b128(la, ga);
      async_b128(lb, gb);
    }
  };

  v8f acc[4][2];
#pragma unroll
  for (int i = 0; i < 4; i++)
#pragma unroll
    for (int j = 0; j < 2; j++) acc[i][j] = v8f{};

  // Prologue: stage tile 0.
  issue_tile(0, 0);
  wait_async0();
  __syncthreads();

  const int nk = K / BK;
  for (int it = 0; it < nk; ++it) {
    const int cur = it & 1;
    if (it + 1 < nk) issue_tile((it + 1) * BK, cur ^ 1);  // overlap with WMMAs

    // Preload ALL fragments of this K-stage (8 A + 4 B = 96 VGPRs), then run
    // the 16 WMMAs as one burst with a single dscnt wait in front.
    FragBF a[2][4], b[2][2];
#pragma unroll
    for (int kk2 = 0; kk2 < 2; kk2++) {
      const int kk = kk2 * 32;
      // 16-bit A layout: lane (r,hi) holds row r, K = kk+hi*8..+7 and kk+16+hi*8..+7
#pragma unroll
      for (int i = 0; i < 4; i++) {
        const unsigned short* p = &sA[cur][(wm * 64 + i * 16 + r) * LDA + kk + hi * 8];
        a[kk2][i].q[0] = *(const uint4*)(p);
        a[kk2][i].q[1] = *(const uint4*)(p + 16);
      }
#pragma unroll
      for (int j = 0; j < 2; j++) {
        const unsigned short* p = &sB[cur][(wn * 32 + j * 16 + r) * LDA + kk + hi * 8];
        b[kk2][j].q[0] = *(const uint4*)(p);
        b[kk2][j].q[1] = *(const uint4*)(p + 16);
      }
    }
#pragma unroll
    for (int kk2 = 0; kk2 < 2; kk2++)
#pragma unroll
      for (int i = 0; i < 4; i++)
#pragma unroll
        for (int j = 0; j < 2; j++)
          acc[i][j] = __builtin_amdgcn_wmma_f32_16x16x32_bf16(
              false, a[kk2][i].v, false, b[kk2][j].v, (short)0, acc[i][j],
              false, false);

    wait_async0();    // next tile fully landed in LDS
    __syncthreads();  // all waves done reading sX[cur] before it is reused
  }

  // Epilogue. f32 C/D layout: VGPR v, lane (r,hi) -> row = v + hi*8, col = r.
#pragma unroll
  for (int i = 0; i < 4; i++) {
#pragma unroll
    for (int j = 0; j < 2; j++) {
      int row0 = bm + wm * 64 + i * 16 + hi * 8;
      int col  = bn + wn * 32 + j * 16 + r;
#pragma unroll
      for (int v = 0; v < 8; v++) {
        int   row = row0 + v;
        float val = acc[i][j][v];
        if constexpr (EPI == EPI_F32) {
          ((float*)C)[(size_t)row * N + col] = val;
        } else if constexpr (EPI == EPI_BF16T) {
          ((unsigned short*)C)[(size_t)col * ldt + row] = f2bf(val);
        } else if constexpr (EPI == EPI_GELU) {
          float x = val + bias[col];
          float g = 0.5f * x * (1.0f + erff(x * 0.70710678118654752f));
          ((unsigned short*)C)[(size_t)row * N + col] = f2bf(g);
        } else {  // EPI_FINAL
          float  x   = val + bias[col];
          float  s   = 1.0f / (1.0f + expf(-x));
          size_t idx = (size_t)row * N + col;
          ((float*)C)[idx] = hsrc[idx] + s * csrc[idx];
        }
      }
    }
  }
}

// ---------------------------------------------------------------------------
// Row L2-normalize: nrm[row,:] = proj[row,:] / max(||proj[row,:]||, 1e-12) -> bf16
// ---------------------------------------------------------------------------
__global__ __launch_bounds__(256)
void rownorm_kernel(const float* __restrict__ proj, unsigned short* __restrict__ nrm) {
  __shared__ float red[256];
  int row = blockIdx.x, tid = threadIdx.x;
  const float* p = proj + (size_t)row * H_DIM;
  float vals[16];
  float ss = 0.f;
#pragma unroll
  for (int t = 0; t < 16; t++) {
    float v = p[tid + t * 256];
    vals[t] = v;
    ss += v * v;
  }
  red[tid] = ss;
  __syncthreads();
  for (int s = 128; s > 0; s >>= 1) {
    if (tid < s) red[tid] += red[tid + s];
    __syncthreads();
  }
  float inv = 1.0f / fmaxf(sqrtf(red[0]), 1e-12f);
  unsigned short* o = nrm + (size_t)row * H_DIM;
#pragma unroll
  for (int t = 0; t < 16; t++) o[tid + t * 256] = f2bf(vals[t] * inv);
}

// ---------------------------------------------------------------------------
// Per-row: diag/attn mask -> top-8 threshold -> softmax -> bf16 weights
// ---------------------------------------------------------------------------
__global__ __launch_bounds__(256)
void topk_softmax_kernel(const float* __restrict__ sim,
                         const unsigned char* __restrict__ mask,
                         unsigned short* __restrict__ weights) {
  __shared__ float cand[256 * 8];
  __shared__ float red[256];
  __shared__ float s_kth, s_max;
  int row = blockIdx.x, tid = threadIdx.x;
  const float NEG = -__builtin_inff();
  const float* p = sim + (size_t)row * B_DIM;

  float ls[16];
  float t8[8];
#pragma unroll
  for (int k = 0; k < 8; k++) t8[k] = NEG;
#pragma unroll
  for (int t = 0; t < 16; t++) {
    int   c = tid + t * 256;
    float v = p[c];
    if (c == row || !mask[c]) v = NEG;
    ls[t] = v;
    if (v > t8[7]) {  // insert into sorted-descending top-8
      t8[7] = v;
#pragma unroll
      for (int k = 7; k > 0; k--)
        if (t8[k] > t8[k - 1]) { float tm = t8[k - 1]; t8[k - 1] = t8[k]; t8[k] = tm; }
    }
  }
#pragma unroll
  for (int k = 0; k < 8; k++) cand[tid * 8 + k] = t8[k];
  __syncthreads();

  if (tid == 0) {  // row top-8 is a subset of the 2048 per-thread candidates
    float g8[8];
#pragma unroll
    for (int k = 0; k < 8; k++) g8[k] = NEG;
    for (int i = 0; i < 2048; i++) {
      float v = cand[i];
      if (v > g8[7]) {
        g8[7] = v;
#pragma unroll
        for (int k = 7; k > 0; k--)
          if (g8[k] > g8[k - 1]) { float tm = g8[k - 1]; g8[k - 1] = g8[k]; g8[k] = tm; }
      }
    }
    s_kth = g8[7];
    s_max = g8[0];
  }
  __syncthreads();

  float kth = s_kth, mx = s_max;
  float ew[16];
  float sum = 0.f;
#pragma unroll
  for (int t = 0; t < 16; t++) {
    float v = ls[t];
    float e = (v != NEG && v >= kth) ? expf(v - mx) : 0.f;
    ew[t] = e;
    sum += e;
  }
  red[tid] = sum;
  __syncthreads();
  for (int s = 128; s > 0; s >>= 1) {
    if (tid < s) red[tid] += red[tid + s];
    __syncthreads();
  }
  float inv = red[0] > 0.f ? 1.0f / red[0] : 0.f;  // all -inf row -> zeros
  unsigned short* o = weights + (size_t)row * B_DIM;
#pragma unroll
  for (int t = 0; t < 16; t++) o[tid + t * 256] = f2bf(ew[t] * inv);
}

// ---------------------------------------------------------------------------
// Dual LayerNorm -> gate_in[B,3H] = [ln_h | ln_a | ln_h*ln_a] in bf16
// ---------------------------------------------------------------------------
__global__ __launch_bounds__(256)
void ln_gate_kernel(const float* __restrict__ hid, const float* __restrict__ cross,
                    const float* __restrict__ g_h, const float* __restrict__ b_h,
                    const float* __restrict__ g_a, const float* __restrict__ b_a,
                    unsigned short* __restrict__ gatein) {
  __shared__ float r0[256], r1[256], r2[256], r3[256];
  int row = blockIdx.x, tid = threadIdx.x;
  const float* ph = hid + (size_t)row * H_DIM;
  const float* pa = cross + (size_t)row * H_DIM;
  float hv[16], av[16];
  float sh = 0.f, sh2 = 0.f, sa = 0.f, sa2 = 0.f;
#pragma unroll
  for (int t = 0; t < 16; t++) {
    int   c = tid + t * 256;
    float h = ph[c], a = pa[c];
    hv[t] = h; av[t] = a;
    sh += h; sh2 += h * h;
    sa += a; sa2 += a * a;
  }
  r0[tid] = sh; r1[tid] = sh2; r2[tid] = sa; r3[tid] = sa2;
  __syncthreads();
  for (int s = 128; s > 0; s >>= 1) {
    if (tid < s) {
      r0[tid] += r0[tid + s]; r1[tid] += r1[tid + s];
      r2[tid] += r2[tid + s]; r3[tid] += r3[tid + s];
    }
    __syncthreads();
  }
  const float invH = 1.0f / (float)H_DIM;
  float muh = r0[0] * invH, varh = r1[0] * invH - muh * muh;
  float mua = r2[0] * invH, vara = r3[0] * invH - mua * mua;
  float rh = rsqrtf(varh + 1e-5f), ra = rsqrtf(vara + 1e-5f);
  unsigned short* o = gatein + (size_t)row * (size_t)K3H;
#pragma unroll
  for (int t = 0; t < 16; t++) {
    int   c  = tid + t * 256;
    float lh = (hv[t] - muh) * rh * g_h[c] + b_h[c];
    float la = (av[t] - mua) * ra * g_a[c] + b_a[c];
    o[c]             = f2bf(lh);
    o[H_DIM + c]     = f2bf(la);
    o[2 * H_DIM + c] = f2bf(lh * la);
  }
}

// ---------------------------------------------------------------------------
// Host-side orchestration
// ---------------------------------------------------------------------------
extern "C" void kernel_launch(void* const* d_in, const int* in_sizes, int n_in,
                              void* d_out, int out_size, void* d_ws, size_t ws_size,
                              hipStream_t stream) {
  const float*         hidden = (const float*)d_in[0];
  const unsigned char* mask   = (const unsigned char*)d_in[1];
  const float*         Ws     = (const float*)d_in[2];
  const float*         Wv     = (const float*)d_in[3];
  const float*         g_h    = (const float*)d_in[4];
  const float*         b_h    = (const float*)d_in[5];
  const float*         g_a    = (const float*)d_in[6];
  const float*         b_a    = (const float*)d_in[7];
  const float*         W1     = (const float*)d_in[8];
  const float*         b1     = (const float*)d_in[9];
  const float*         W2     = (const float*)d_in[10];
  const float*         b2     = (const float*)d_in[11];

  char*        ws = (char*)d_ws;
  const size_t MB = 1ull << 20;
  // Bump allocation with lifetime-based reuse (~384 MB peak):
  unsigned short* hbf  = (unsigned short*)(ws + 0);        // hidden bf16       32MB
  unsigned short* wbuf = (unsigned short*)(ws + 32 * MB);  // Ws/Wv/W1/W2 bf16  32MB
  float*          bigC = (float*)(ws + 64 * MB);           // proj then cross   64MB
  unsigned short* nbuf = (unsigned short*)(ws + 128 * MB); // nrm then hidden1  32MB
  float*          simf = (float*)(ws + 160 * MB);          // sim               64MB
  unsigned short* wts  = (unsigned short*)(ws + 224 * MB); // weights bf16      32MB
  unsigned short* valT = (unsigned short*)(ws + 256 * MB); // values^T bf16     32MB
  unsigned short* gin  = (unsigned short*)(ws + 288 * MB); // gate_in bf16      96MB

  dim3 blk(256);
  auto cvt = [&](const float* src, unsigned short* dst, long n) {
    long n4 = n / 4;
    cvt_bf16_kernel<<<dim3((unsigned)((n4 + 255) / 256)), blk, 0, stream>>>(src, dst, n4);
  };

  // 1) proj = hidden @ Ws^T
  cvt(hidden, hbf, (long)B_DIM * H_DIM);
  cvt(Ws, wbuf, (long)H_DIM * H_DIM);
  gemm_bf16_nt<EPI_F32><<<dim3(H_DIM / 128, B_DIM / 128), blk, 0, stream>>>(
      hbf, wbuf, bigC, B_DIM, H_DIM, H_DIM, nullptr, nullptr, nullptr, 0);

  // 2) nrm = L2-normalize(proj);  sim = nrm @ nrm^T  (TEMP = 1)
  rownorm_kernel<<<dim3(B_DIM), blk, 0, stream>>>(bigC, nbuf);
  gemm_bf16_nt<EPI_F32><<<dim3(B_DIM / 128, B_DIM / 128), blk, 0, stream>>>(
      nbuf, nbuf, simf, B_DIM, B_DIM, H_DIM, nullptr, nullptr, nullptr, 0);

  // 3) mask + top-8 threshold + softmax -> weights (bf16)
  topk_softmax_kernel<<<dim3(B_DIM), blk, 0, stream>>>(simf, mask, wts);

  // 4) values = hidden @ Wv^T, stored transposed as valT[H,B] bf16
  cvt(Wv, wbuf, (long)H_DIM * H_DIM);
  gemm_bf16_nt<EPI_BF16T><<<dim3(H_DIM / 128, B_DIM / 128), blk, 0, stream>>>(
      hbf, wbuf, valT, B_DIM, H_DIM, H_DIM, nullptr, nullptr, nullptr, B_DIM);

  // 5) cross = weights @ values  == weights[B,B] * valT[H,B]^T  (NT form)
  gemm_bf16_nt<EPI_F32><<<dim3(H_DIM / 128, B_DIM / 128), blk, 0, stream>>>(
      wts, valT, bigC, B_DIM, H_DIM, B_DIM, nullptr, nullptr, nullptr, 0);

  // 6) gate_in = [ln_h | ln_a | ln_h*ln_a]  (bf16, [B,3H])
  ln_gate_kernel<<<dim3(B_DIM), blk, 0, stream>>>(hidden, bigC, g_h, b_h, g_a, b_a, gin);

  // 7) hidden1 = gelu(gate_in @ W1^T + b1) -> bf16
  cvt(W1, wbuf, (long)GH_DIM * K3H);
  gemm_bf16_nt<EPI_GELU><<<dim3(GH_DIM / 128, B_DIM / 128), blk, 0, stream>>>(
      gin, wbuf, nbuf, B_DIM, GH_DIM, K3H, b1, nullptr, nullptr, 0);

  // 8) out = hidden + sigmoid(hidden1 @ W2^T + b2) * cross   (fused epilogue)
  cvt(W2, wbuf, (long)H_DIM * GH_DIM);
  gemm_bf16_nt<EPI_FINAL><<<dim3(H_DIM / 128, B_DIM / 128), blk, 0, stream>>>(
      nbuf, wbuf, d_out, B_DIM, H_DIM, GH_DIM, b2, hidden, bigC, 0);
}
